// PointNet2Encoder_36094905156308
// MI455X (gfx1250) — matrix operand order repaired and checked by reading, not compiled, not run on hardware
//
#include <hip/hip_runtime.h>

typedef __attribute__((ext_vector_type(16))) _Float16 v16h;
typedef __attribute__((ext_vector_type(8)))  _Float16 v8h;
typedef __attribute__((ext_vector_type(8)))  float    v8f;
typedef __attribute__((ext_vector_type(4)))  int      v4i;

// ---------------------------------------------------------------------------
// Optional CDNA5 async global->LDS copy (ASYNCcnt-tracked). Falls back to a
// plain vector copy if the builtins are not declared by this toolchain.
// Probe-verified signature: (v4i addrspace(1)* src, v4i addrspace(3)* dst,
//                            imm offset, imm cpol).
// ---------------------------------------------------------------------------
#if __has_builtin(__builtin_amdgcn_global_load_async_to_lds_b128) && \
    __has_builtin(__builtin_amdgcn_s_wait_asynccnt)
#define HAVE_ASYNC_LDS 1
#else
#define HAVE_ASYNC_LDS 0
#endif

__device__ inline void copy16_g2l(void* lds_dst, const void* gsrc) {
#if HAVE_ASYNC_LDS
    __builtin_amdgcn_global_load_async_to_lds_b128(
        (__attribute__((address_space(1))) v4i*)gsrc,
        (__attribute__((address_space(3))) v4i*)lds_dst, 0, 0);
#else
    *(v8h*)lds_dst = *(const v8h*)gsrc;
#endif
}

__device__ inline void copy_wait() {
#if HAVE_ASYNC_LDS
    __builtin_amdgcn_s_wait_asynccnt(0);
#endif
}

// ---------------------------------------------------------------------------
// WMMA fragment loaders (CDNA5 16x16x32 f16 layouts, ISA 7.12.2).
// A (16x32 f16): lane l(0-15) row M=l, k = {base+0..7, base+16..23} with
// base = kt*32 + 8*(lane>>4). Two contiguous 8-half chunks -> two b128 loads.
// ---------------------------------------------------------------------------
__device__ inline v16h load_a_frag(const _Float16* p) {
    v8h lo = *(const v8h*)(p);
    v8h hi = *(const v8h*)(p + 16);
    return __builtin_shufflevector(lo, hi, 0, 1, 2, 3, 4, 5, 6, 7,
                                   8, 9, 10, 11, 12, 13, 14, 15);
}

// B fragment from pre-packed weights: per lane 16 contiguous halfs.
__device__ inline v16h load_b_packed(const _Float16* wp) {
    v8h lo = *(const v8h*)(wp);
    v8h hi = *(const v8h*)(wp + 8);
    return __builtin_shufflevector(lo, hi, 0, 1, 2, 3, 4, 5, 6, 7,
                                   8, 9, 10, 11, 12, 13, 14, 15);
}

// ---------------------------------------------------------------------------
// Pack f32 weights (Kact x Nout, row-major) into WMMA-B fragment layout:
// Wp[((nt*(Kpad/32)+kt)*32 + lane)*16 + e] = W[(k+rot)%Kact, col], 0 if k>=Kact.
// rot=3 absorbs the [feats|rel] column reordering of layer-0 inputs.
// ---------------------------------------------------------------------------
__global__ void pack_w_kernel(const float* __restrict__ W, int Kact, int Kpad,
                              int Nout, int rot, _Float16* __restrict__ Wp) {
    int q = blockIdx.x * blockDim.x + threadIdx.x;
    if (q >= Kpad * Nout) return;
    int e    = q & 15;
    int lane = (q >> 4) & 31;
    int rest = q >> 9;
    int nkt  = Kpad >> 5;
    int kt   = rest % nkt;
    int nt   = rest / nkt;
    int k    = kt * 32 + (lane >> 4) * 8 + (e < 8 ? e : e + 8);
    int col  = nt * 16 + (lane & 15);
    _Float16 v = (_Float16)0.f;
    if (k < Kact) {
        int src = (k + rot) % Kact;
        v = (_Float16)W[(size_t)src * Nout + col];
    }
    Wp[q] = v;
}

// ---------------------------------------------------------------------------
// One MLP layer: O[KPTS x Nout] = relu(X[KPTS x Kpad] @ W + b).
// B fragment loaded once per k-tile, reused across all KPTS/16 row tiles.
// ---------------------------------------------------------------------------
template <int KPTS>
__device__ inline void mlp_layer(const _Float16* Xl, int ldx, int Kpad,
                                 const _Float16* Wp, const float* bias, int Nout,
                                 _Float16* Ol, int nw) {
    constexpr int MT = KPTS / 16;
    int wid = threadIdx.x >> 5, lane = threadIdx.x & 31;
    int l = lane & 15, h = lane >> 4;
    int nkt = Kpad >> 5;
    for (int nt = wid; nt < Nout / 16; nt += nw) {
        float bc = bias[nt * 16 + l];
        v8f acc[MT] = {};
        const _Float16* wp = Wp + ((size_t)nt * nkt * 32 + lane) * 16;
        for (int kt = 0; kt < nkt; ++kt) {
            v16h bf = load_b_packed(wp);
            wp += 32 * 16;
            if (kt + 1 < nkt) __builtin_prefetch(wp, 0, 3);
#pragma unroll
            for (int m = 0; m < MT; ++m) {
                v16h a = load_a_frag(Xl + (size_t)(m * 16 + l) * ldx + kt * 32 + h * 8);
                acc[m] = __builtin_amdgcn_wmma_f32_16x16x32_f16(false, a, false, bf,
                                                                (short)0, acc[m],
                                                                false, false);
            }
        }
        int col = nt * 16 + l;
#pragma unroll
        for (int m = 0; m < MT; ++m)
#pragma unroll
            for (int v = 0; v < 8; ++v) {
                int row = m * 16 + v + h * 8;  // C layout: lanes 0-15 M=v, 16-31 M=v+8
                Ol[(size_t)row * Nout + col] = (_Float16)fmaxf(acc[m][v] + bc, 0.f);
            }
    }
}

// Final layer fused with column-max over the KPTS rows:
// out[col] = relu(max_m(X@W)[m,col] + b[col])   (bias/relu commute with max)
template <int KPTS>
__device__ inline void mlp_layer_max(const _Float16* Xl, int ldx, int Kpad,
                                     const _Float16* Wp, const float* bias, int Nout,
                                     _Float16* Og, int nw) {
    constexpr int MT = KPTS / 16;
    int wid = threadIdx.x >> 5, lane = threadIdx.x & 31;
    int l = lane & 15, h = lane >> 4;
    int nkt = Kpad >> 5;
    for (int nt = wid; nt < Nout / 16; nt += nw) {
        int col = nt * 16 + l;
        v8f acc[MT] = {};
        const _Float16* wp = Wp + ((size_t)nt * nkt * 32 + lane) * 16;
        for (int kt = 0; kt < nkt; ++kt) {
            v16h bf = load_b_packed(wp);
            wp += 32 * 16;
            if (kt + 1 < nkt) __builtin_prefetch(wp, 0, 3);
#pragma unroll
            for (int m = 0; m < MT; ++m) {
                v16h a = load_a_frag(Xl + (size_t)(m * 16 + l) * ldx + kt * 32 + h * 8);
                acc[m] = __builtin_amdgcn_wmma_f32_16x16x32_f16(false, a, false, bf,
                                                                (short)0, acc[m],
                                                                false, false);
            }
        }
        float cmax = -3.0e38f;
#pragma unroll
        for (int m = 0; m < MT; ++m)
#pragma unroll
            for (int v = 0; v < 8; ++v) cmax = fmaxf(cmax, acc[m][v]);
        cmax = fmaxf(cmax, __shfl_xor(cmax, 16, 32));  // merge lane halves (M 0-7 vs 8-15)
        if (lane < 16) Og[col] = (_Float16)fmaxf(cmax + bias[col], 0.f);
    }
}

// ---------------------------------------------------------------------------
// Farthest point sampling: one block per batch; dist in LDS; wave-shuffle
// argmax butterfly + single cross-wave combine (3 barriers per iteration).
// ---------------------------------------------------------------------------
__global__ __launch_bounds__(1024) void fps_kernel(const float* __restrict__ pts,
                                                   int N, int M, float* __restrict__ ctrs) {
    __shared__ float dist[8192];
    __shared__ float rv[32];
    __shared__ int   ri[32];
    __shared__ float lp[3];
    __shared__ int   li;
    int b = blockIdx.x, tid = threadIdx.x, nt = blockDim.x;
    int wid = tid >> 5, lane = tid & 31;
    const float* P = pts + (size_t)b * N * 3;
    for (int i = tid; i < N; i += nt) dist[i] = 1e10f;
    if (tid == 0) li = 0;
    __syncthreads();
    for (int it = 0; it < M; ++it) {
        if (tid < 3) {
            float v = P[(size_t)li * 3 + tid];
            lp[tid] = v;
            ctrs[((size_t)b * M + it) * 3 + tid] = v;
        }
        __syncthreads();
        float l0 = lp[0], l1 = lp[1], l2 = lp[2];
        float bm = -3.0e38f; int bi = 0x7fffffff;
        for (int i = tid; i < N; i += nt) {
            float dx = P[i * 3 + 0] - l0, dy = P[i * 3 + 1] - l1, dz = P[i * 3 + 2] - l2;
            float d  = fminf(dist[i], dx * dx + dy * dy + dz * dz);
            dist[i]  = d;
            if (d > bm || (d == bm && i < bi)) { bm = d; bi = i; }
        }
#pragma unroll
        for (int s = 16; s > 0; s >>= 1) {  // intra-wave argmax butterfly
            float ov = __shfl_xor(bm, s, 32);
            int   oi = __shfl_xor(bi, s, 32);
            if (ov > bm || (ov == bm && oi < bi)) { bm = ov; bi = oi; }
        }
        if (lane == 0) { rv[wid] = bm; ri[wid] = bi; }
        __syncthreads();
        if (wid == 0) {
            float v = rv[lane]; int i = ri[lane];
#pragma unroll
            for (int s = 16; s > 0; s >>= 1) {
                float ov = __shfl_xor(v, s, 32);
                int   oi = __shfl_xor(i, s, 32);
                if (ov > v || (ov == v && oi < i)) { v = ov; i = oi; }
            }
            if (lane == 0) li = i;
        }
        __syncthreads();
    }
}

// ---------------------------------------------------------------------------
// Ball grouping: thread per center; first K indices (ascending) within r^2,
// padded with the first hit (matches top_k(-index) semantics).
// ---------------------------------------------------------------------------
__global__ void ball_group_kernel(const float* __restrict__ pts,
                                  const float* __restrict__ ctrs,
                                  int B, int N, int M, float r2, int K,
                                  int* __restrict__ idx) {
    int t = blockIdx.x * blockDim.x + threadIdx.x;
    if (t >= B * M) return;
    int b = t / M;
    const float* P = pts + (size_t)b * N * 3;
    float cx = ctrs[(size_t)t * 3 + 0];
    float cy = ctrs[(size_t)t * 3 + 1];
    float cz = ctrs[(size_t)t * 3 + 2];
    int* row = idx + (size_t)t * K;
    int cnt = 0, first = 0;
    for (int i = 0; i < N && cnt < K; ++i) {
        float dx = P[i * 3 + 0] - cx, dy = P[i * 3 + 1] - cy, dz = P[i * 3 + 2] - cz;
        if (dx * dx + dy * dy + dz * dz <= r2) {
            if (cnt == 0) first = i;
            row[cnt++] = i;
        }
    }
    for (int j = cnt; j < K; ++j) row[j] = first;
}

// ---------------------------------------------------------------------------
// Fused set-abstraction MLP: one workgroup per ball group.
// X layout per row: [feats(F) | rel_xyz(3) | 0-pad]; layer-0 weights are
// pre-rotated by 3 in pack_w_kernel to match. Feats rows (16B-aligned) are
// gathered with async global->LDS b128 copies when available.
// ---------------------------------------------------------------------------
template <int KPTS, int INPAD, int F, int H0, int H1, int OUT, int NW>
__global__ __launch_bounds__(NW * 32) void sa_mlp_kernel(
    const float* __restrict__ pts, const _Float16* __restrict__ feats,
    const float* __restrict__ ctrs, const int* __restrict__ idx,
    const _Float16* pw0, const float* b0, const _Float16* pw1, const float* b1,
    const _Float16* pw2, const float* b2, _Float16* __restrict__ outf,
    int N, int M) {
    __shared__ __attribute__((aligned(16))) _Float16 X[KPTS * INPAD];
    __shared__ __attribute__((aligned(16))) _Float16 Hb0[KPTS * H0];
    __shared__ __attribute__((aligned(16))) _Float16 Hb1[KPTS * H1];
    __shared__ int   sidx[KPTS];
    __shared__ float sc[3];
    int g = blockIdx.x;
    int b = g / M;
    int tid = threadIdx.x;
    if (tid < KPTS) sidx[tid] = idx[(size_t)g * KPTS + tid];
    if (tid < 3)    sc[tid]   = ctrs[(size_t)g * 3 + tid];
    __syncthreads();
    if (F > 0) {  // gather features: KPTS rows x F halfs as 16B chunks
        constexpr int CH = (F > 0) ? F / 8 : 1;
        for (int e = tid; e < KPTS * CH; e += NW * 32) {
            int j = e / CH, q = e % CH;
            int ij = sidx[j];
            copy16_g2l(&X[j * INPAD + q * 8],
                       &feats[((size_t)b * N + ij) * F + q * 8]);
        }
    }
    for (int e = tid; e < KPTS * (INPAD - F); e += NW * 32) {  // rel xyz + pad
        int j = e / (INPAD - F), c0 = e % (INPAD - F);
        int ij = sidx[j];
        float v = (c0 < 3) ? pts[((size_t)b * N + ij) * 3 + c0] - sc[c0] : 0.f;
        X[j * INPAD + F + c0] = (_Float16)v;
    }
    copy_wait();
    __syncthreads();
    mlp_layer<KPTS>(X, INPAD, INPAD, pw0, b0, H0, Hb0, NW);
    __syncthreads();
    mlp_layer<KPTS>(Hb0, H0, H0, pw1, b1, H1, Hb1, NW);
    __syncthreads();
    mlp_layer_max<KPTS>(Hb1, H1, H1, pw2, b2, OUT, outf + (size_t)g * OUT, NW);
}

// ---------------------------------------------------------------------------
// Global GEMM + bias + relu for SA3 (A f16 MxK row-major, packed W, C f16).
// Block = 4 waves; wave computes 64x16 (4 row tiles); grid = (M/64, N/64).
// ---------------------------------------------------------------------------
__global__ __launch_bounds__(128) void gemm_bias_relu_kernel(
    const _Float16* __restrict__ A, const _Float16* __restrict__ Wp,
    const float* __restrict__ bias, _Float16* __restrict__ C,
    int Kpad, int Nout) {
    int wid = threadIdx.x >> 5, lane = threadIdx.x & 31;
    int l = lane & 15, h = lane >> 4;
    int mt0 = blockIdx.x * 4;
    int nt  = blockIdx.y * 4 + wid;
    int col = nt * 16 + l;
    int nkt = Kpad >> 5;
    v8f acc[4] = {};
    const _Float16* wp = Wp + ((size_t)nt * nkt * 32 + lane) * 16;
    for (int kt = 0; kt < nkt; ++kt) {
        v16h bf = load_b_packed(wp);
        wp += 32 * 16;
        if (kt + 1 < nkt) __builtin_prefetch(wp, 0, 3);
#pragma unroll
        for (int m = 0; m < 4; ++m) {
            v16h a = load_a_frag(A + (size_t)((mt0 + m) * 16 + l) * Kpad + kt * 32 + h * 8);
            acc[m] = __builtin_amdgcn_wmma_f32_16x16x32_f16(false, a, false, bf,
                                                            (short)0, acc[m],
                                                            false, false);
        }
    }
    float bc = bias[col];
#pragma unroll
    for (int m = 0; m < 4; ++m)
#pragma unroll
        for (int v = 0; v < 8; ++v) {
            int row = (mt0 + m) * 16 + v + h * 8;
            C[(size_t)row * Nout + col] = (_Float16)fmaxf(acc[m][v] + bc, 0.f);
        }
}

// Build SA3 input rows (16*128): [f2(256) | c2(3) | 0-pad(29)] (rot-3 packed W).
__global__ void build_a3_kernel(const float* __restrict__ c2,
                                const _Float16* __restrict__ f2,
                                _Float16* __restrict__ A3) {
    int t = blockIdx.x * blockDim.x + threadIdx.x;
    if (t >= 2048 * 36) return;           // 36 chunks of 8 halfs per 288-col row
    int r = t / 36, q = t % 36;
    if (q < 32) {
        *(v8h*)&A3[(size_t)r * 288 + q * 8] = *(const v8h*)&f2[(size_t)r * 256 + q * 8];
    } else {
        int c0 = (q - 32) * 8;
        v8h v = {};
#pragma unroll
        for (int i = 0; i < 8; ++i) {
            int c = c0 + i;
            if (c < 3) v[i] = (_Float16)c2[(size_t)r * 3 + c];
        }
        *(v8h*)&A3[(size_t)r * 288 + 256 + c0] = v;
    }
}

// out[b][n] = max over 128 rows of h2 (relu already applied); f32 output.
__global__ void reduce_max_kernel(const _Float16* __restrict__ h2,
                                  float* __restrict__ out) {
    int t = blockIdx.x * blockDim.x + threadIdx.x;
    if (t >= 16 * 1024) return;
    int b = t / 1024, n = t % 1024;
    float mx = -3.0e38f;
    for (int j = 0; j < 128; ++j)
        mx = fmaxf(mx, (float)h2[((size_t)(b * 128 + j)) * 1024 + n]);
    out[t] = mx;
}

// ---------------------------------------------------------------------------
extern "C" void kernel_launch(void* const* d_in, const int* in_sizes, int n_in,
                              void* d_out, int out_size, void* d_ws, size_t ws_size,
                              hipStream_t stream) {
    const float* x    = (const float*)d_in[0];
    const float* s1w0 = (const float*)d_in[1];  const float* s1b0 = (const float*)d_in[2];
    const float* s1w1 = (const float*)d_in[3];  const float* s1b1 = (const float*)d_in[4];
    const float* s1w2 = (const float*)d_in[5];  const float* s1b2 = (const float*)d_in[6];
    const float* s2w0 = (const float*)d_in[7];  const float* s2b0 = (const float*)d_in[8];
    const float* s2w1 = (const float*)d_in[9];  const float* s2b1 = (const float*)d_in[10];
    const float* s2w2 = (const float*)d_in[11]; const float* s2b2 = (const float*)d_in[12];
    const float* s3w0 = (const float*)d_in[13]; const float* s3b0 = (const float*)d_in[14];
    const float* s3w1 = (const float*)d_in[15]; const float* s3b1 = (const float*)d_in[16];
    const float* s3w2 = (const float*)d_in[17]; const float* s3b2 = (const float*)d_in[18];
    float* out = (float*)d_out;

    char* ws = (char*)d_ws;
    size_t off = 0;
    auto alloc = [&](size_t bytes) -> void* {
        void* p = ws + off;
        off = (off + bytes + 255) & ~(size_t)255;
        return p;
    };
    float*    c1   = (float*)alloc((size_t)16 * 512 * 3 * sizeof(float));
    int*      idx1 = (int*)alloc((size_t)16 * 512 * 32 * sizeof(int));
    _Float16* f1   = (_Float16*)alloc((size_t)16 * 512 * 128 * sizeof(_Float16));
    float*    c2   = (float*)alloc((size_t)16 * 128 * 3 * sizeof(float));
    int*      idx2 = (int*)alloc((size_t)16 * 128 * 64 * sizeof(int));
    _Float16* f2   = (_Float16*)alloc((size_t)16 * 128 * 256 * sizeof(_Float16));
    _Float16* A3   = (_Float16*)alloc((size_t)2048 * 288 * sizeof(_Float16));
    _Float16* h0   = (_Float16*)alloc((size_t)2048 * 256 * sizeof(_Float16));
    _Float16* h1   = (_Float16*)alloc((size_t)2048 * 512 * sizeof(_Float16));
    _Float16* h2   = (_Float16*)alloc((size_t)2048 * 1024 * sizeof(_Float16));
    // Packed f16 weight fragments (zero-padded to Kpad).
    _Float16* p1w0 = (_Float16*)alloc((size_t)32 * 64 * 2);
    _Float16* p1w1 = (_Float16*)alloc((size_t)64 * 64 * 2);
    _Float16* p1w2 = (_Float16*)alloc((size_t)64 * 128 * 2);
    _Float16* p2w0 = (_Float16*)alloc((size_t)160 * 128 * 2);
    _Float16* p2w1 = (_Float16*)alloc((size_t)128 * 128 * 2);
    _Float16* p2w2 = (_Float16*)alloc((size_t)128 * 256 * 2);
    _Float16* p3w0 = (_Float16*)alloc((size_t)288 * 256 * 2);
    _Float16* p3w1 = (_Float16*)alloc((size_t)256 * 512 * 2);
    _Float16* p3w2 = (_Float16*)alloc((size_t)512 * 1024 * 2);

    auto packw = [&](const float* W, int Kact, int Kpad, int Nout, int rot,
                     _Float16* Wp) {
        int total = Kpad * Nout;
        pack_w_kernel<<<(total + 255) / 256, 256, 0, stream>>>(W, Kact, Kpad, Nout,
                                                               rot, Wp);
    };
    packw(s1w0, 3, 32, 64, 3, p1w0);      packw(s1w1, 64, 64, 64, 0, p1w1);
    packw(s1w2, 64, 64, 128, 0, p1w2);    packw(s2w0, 131, 160, 128, 3, p2w0);
    packw(s2w1, 128, 128, 128, 0, p2w1);  packw(s2w2, 128, 128, 256, 0, p2w2);
    packw(s3w0, 259, 288, 256, 3, p3w0);  packw(s3w1, 256, 256, 512, 0, p3w1);
    packw(s3w2, 512, 512, 1024, 0, p3w2);

    // ---- SA1: FPS 8192->512, ball r=0.2 k=32, MLP 3->64->64->128, max ----
    fps_kernel<<<16, 1024, 0, stream>>>(x, 8192, 512, c1);
    ball_group_kernel<<<(16 * 512 + 255) / 256, 256, 0, stream>>>(
        x, c1, 16, 8192, 512, 0.04f, 32, idx1);
    sa_mlp_kernel<32, 32, 0, 64, 64, 128, 4><<<16 * 512, 128, 0, stream>>>(
        x, (const _Float16*)nullptr, c1, idx1,
        p1w0, s1b0, p1w1, s1b1, p1w2, s1b2, f1, 8192, 512);

    // ---- SA2: FPS 512->128, ball r=0.4 k=64, MLP 131->128->128->256, max ----
    fps_kernel<<<16, 1024, 0, stream>>>(c1, 512, 128, c2);
    ball_group_kernel<<<(16 * 128 + 255) / 256, 256, 0, stream>>>(
        c1, c2, 16, 512, 128, 0.16f, 64, idx2);
    sa_mlp_kernel<64, 160, 128, 128, 128, 256, 4><<<16 * 128, 128, 0, stream>>>(
        c1, f1, c2, idx2,
        p2w0, s2b0, p2w1, s2b1, p2w2, s2b2, f2, 512, 128);

    // ---- SA3: [f2|c2] -> MLP 259->256->512->1024 -> max over 128 ----
    build_a3_kernel<<<(2048 * 36 + 255) / 256, 256, 0, stream>>>(c2, f2, A3);
    {
        dim3 g0(2048 / 64, 256 / 64);
        gemm_bias_relu_kernel<<<g0, 128, 0, stream>>>(A3, p3w0, s3b0, h0, 288, 256);
        dim3 g1(2048 / 64, 512 / 64);
        gemm_bias_relu_kernel<<<g1, 128, 0, stream>>>(h0, p3w1, s3b1, h1, 256, 512);
        dim3 g2(2048 / 64, 1024 / 64);
        gemm_bias_relu_kernel<<<g2, 128, 0, stream>>>(h1, p3w2, s3b2, h2, 512, 1024);
    }
    reduce_max_kernel<<<(16 * 1024 + 255) / 256, 256, 0, stream>>>(h2, out);

    (void)in_sizes; (void)n_in; (void)out_size; (void)ws_size;
}
